// TRL_80461917323569
// MI455X (gfx1250) — compile-verified, matrix operand order checked by reading, not run on hardware
//
#include <hip/hip_runtime.h>
#include <hip/hip_bf16.h>

// ---------------------------------------------------------------------------
// 6-layer transformer encoder for MI455X (gfx1250, wave32, WMMA).
// All GEMMs run on v_wmma_f32_16x16x32_bf16 (bf16 in, fp32 accumulate).
// GEMM: 128x128 block tile, 8 waves, 32x64 per wave (8 WMMA / k-step / wave),
// double-buffered LDS staging (1 barrier per k-step).
// Weights converted+transposed to bf16 once per launch; activations converted
// fp32->bf16 during LDS staging. Temporal bias computed once per layer and
// shared across heads. 1/sqrt(dk) folded into Q.
// ---------------------------------------------------------------------------

typedef __attribute__((ext_vector_type(16))) __bf16 v16bf;
typedef __attribute__((ext_vector_type(8)))  float  v8f;
typedef __attribute__((ext_vector_type(4)))  int    v4i;

union FragU { v16bf bf; v4i i4[2]; };

__device__ __forceinline__ unsigned short f2bf(float x) {
  __bf16 h = (__bf16)x;
  return __builtin_bit_cast(unsigned short, h);
}
__device__ __forceinline__ unsigned pack2bf(float a, float b) {
  return (unsigned)f2bf(a) | ((unsigned)f2bf(b) << 16);
}
__device__ __forceinline__ v8f wmma_bf16(v16bf a, v16bf b, v8f c) {
  return __builtin_amdgcn_wmma_f32_16x16x32_bf16(false, a, false, b, (short)0, c,
                                                 false, false);
}
__device__ __forceinline__ float gelu_f(float x) {
  return 0.5f * x * (1.0f + erff(x * 0.70710678118654752440f));
}

// Load a 16x32 bf16 WMMA operand fragment (A-layout) from an LDS tile stored
// row-major with `stride` bytes/row, rows starting at row0, K offset = kbyte.
// Per ISA: lane<16 holds K0..7 (+0B) and K16..23 (+32B); lane>=16 holds
// K8..15 (+16B) and K24..31 (+48B).
__device__ __forceinline__ v16bf lds_frag(const char* base, int row0, int stride,
                                          int kbyte) {
  int lane = threadIdx.x & 31;
  int lr = lane & 15;
  int hi = (lane >> 4) & 1;
  const char* p = base + (row0 + lr) * stride + kbyte + hi * 16;
  FragU f;
  f.i4[0] = *(const v4i*)p;
  f.i4[1] = *(const v4i*)(p + 32);
  return f.bf;
}

// Stage a 128x32 fp32 tile -> bf16 LDS tile (80 B/row). 256 threads.
__device__ __forceinline__ void stage_f32_128(char* dst, const float* src,
                                              long ld, long r0, long c0) {
  int tid = threadIdx.x;
#pragma unroll
  for (int i = 0; i < 2; ++i) {
    int chunk = tid + i * 256;
    int row = chunk >> 2;
    int col = (chunk & 3) * 8;
    const float* s = src + (r0 + row) * ld + c0 + col;
    float4 f0 = *(const float4*)s;
    float4 f1 = *(const float4*)(s + 4);
    v4i d;
    d.x = (int)pack2bf(f0.x, f0.y);
    d.y = (int)pack2bf(f0.z, f0.w);
    d.z = (int)pack2bf(f1.x, f1.y);
    d.w = (int)pack2bf(f1.z, f1.w);
    *(v4i*)(dst + row * 80 + col * 2) = d;
  }
}

// Stage a 128x32 bf16 tile from global (ld in elements) -> LDS (80 B/row).
__device__ __forceinline__ void stage_bf16_128x32(char* dst,
                                                  const unsigned short* src,
                                                  long ld, long r0, long c0) {
  int tid = threadIdx.x;
#pragma unroll
  for (int i = 0; i < 2; ++i) {
    int chunk = tid + i * 256;
    int row = chunk >> 2;
    int c16 = chunk & 3;
    v4i d = *(const v4i*)(src + (r0 + row) * ld + c0 + c16 * 8);
    *(v4i*)(dst + row * 80 + c16 * 16) = d;
  }
}

// Stage a 64x32 bf16 tile from global (ld in elements), dstride B/row.
__device__ __forceinline__ void stage_bf16_32(char* dst, int dstride,
                                              const unsigned short* src, long ld,
                                              long r0, long c0) {
  int tid = threadIdx.x;
  int row = tid >> 2;
  int c16 = tid & 3;
  v4i d = *(const v4i*)(src + (r0 + row) * ld + c0 + c16 * 8);
  *(v4i*)(dst + row * dstride + c16 * 16) = d;
}

// Stage a 64x64 bf16 tile from global.
__device__ __forceinline__ void stage_bf16_64(char* dst, int dstride,
                                              const unsigned short* src, long ld,
                                              long r0, long c0) {
  int tid = threadIdx.x;
#pragma unroll
  for (int i = 0; i < 2; ++i) {
    int chunk = tid + i * 256;
    int row = chunk >> 3;
    int c16 = chunk & 7;
    v4i d = *(const v4i*)(src + (r0 + row) * ld + c0 + c16 * 8);
    *(v4i*)(dst + row * dstride + c16 * 16) = d;
  }
}

// ---------------------------------------------------------------------------
// Weight convert+transpose: W (K x N fp32, row-major) -> Wt (N x K bf16).
// ---------------------------------------------------------------------------
__global__ void transpose_cvt_kernel(const float* __restrict__ W,
                                     unsigned short* __restrict__ Wt, int K,
                                     int N) {
  __shared__ float tile[32][33];
  long kb = (long)blockIdx.x * 32, nb = (long)blockIdx.y * 32;
  int tx = threadIdx.x & 31, ty = threadIdx.x >> 5;
#pragma unroll
  for (int i = 0; i < 32; i += 8)
    tile[ty + i][tx] = W[(kb + ty + i) * N + nb + tx];
  __syncthreads();
#pragma unroll
  for (int i = 0; i < 32; i += 8)
    Wt[(nb + ty + i) * K + kb + tx] = f2bf(tile[tx][ty + i]);
}

// ---------------------------------------------------------------------------
// Temporal bias: bias(b,t,s) = sum_h tw2[h]*leaky(tb*tw1[h]+tb1[h],0.2)+tb2
// with tb = 1/log(e + tmat). Pointwise; computed once per layer, shared by
// all 8 heads.
// ---------------------------------------------------------------------------
__global__ void bias_mlp_kernel(const float* __restrict__ tmat,
                                const float* __restrict__ tw1,
                                const float* __restrict__ tb1,
                                const float* __restrict__ tw2,
                                const float* __restrict__ tb2,
                                float* __restrict__ out, long n) {
  __shared__ float w1[64], b1[64], w2[64];
  if (threadIdx.x < 64) {
    w1[threadIdx.x] = tw1[threadIdx.x];
    b1[threadIdx.x] = tb1[threadIdx.x];
    w2[threadIdx.x] = tw2[threadIdx.x];
  }
  __syncthreads();
  long i = (long)blockIdx.x * 256 + threadIdx.x;
  if (i >= n) return;
  float tb = 1.0f / logf(2.7182818284590452f + tmat[i]);
  float acc = tb2[0];
#pragma unroll
  for (int h = 0; h < 64; ++h) {
    float z = tb * w1[h] + b1[h];
    z = (z > 0.0f) ? z : 0.2f * z;
    acc += w2[h] * z;
  }
  out[i] = acc;
}

// ---------------------------------------------------------------------------
// GEMM core: 128x128 block, 8 waves (4m x 2n), 32x64 per wave, double-buffered
// LDS. Shared by generic and qkv kernels via a macro-style inline.
// smem layout: sA0 | sA1 | sB0 | sB1, each 128*80 = 10240 B (40960 total).
// ---------------------------------------------------------------------------
struct GemmAcc { v8f a[2][4]; };

__device__ __forceinline__ void gemm_core(GemmAcc& G, char* smem,
                                          const float* A,
                                          const unsigned short* Bt, int K,
                                          long bm, long bn, int wm, int wn) {
  v8f zero = {};
#pragma unroll
  for (int mi = 0; mi < 2; ++mi)
#pragma unroll
    for (int ni = 0; ni < 4; ++ni) G.a[mi][ni] = zero;

  stage_f32_128(smem, A, K, bm * 128, 0);
  stage_bf16_128x32(smem + 20480, Bt, K, bn * 128, 0);
  __syncthreads();
  int cur = 0;
  for (int k0 = 0; k0 < K; k0 += 32) {
    char* sAc = smem + cur * 10240;
    char* sBc = smem + 20480 + cur * 10240;
    if (k0 + 32 < K) {
      stage_f32_128(smem + (cur ^ 1) * 10240, A, K, bm * 128, k0 + 32);
      stage_bf16_128x32(smem + 20480 + (cur ^ 1) * 10240, Bt, K, bn * 128,
                        k0 + 32);
    }
    v16bf a0 = lds_frag(sAc, wm * 32, 80, 0);
    v16bf a1 = lds_frag(sAc, wm * 32 + 16, 80, 0);
#pragma unroll
    for (int ni = 0; ni < 4; ++ni) {
      v16bf b = lds_frag(sBc, wn * 64 + ni * 16, 80, 0);
      G.a[0][ni] = wmma_bf16(a0, b, G.a[0][ni]);
      G.a[1][ni] = wmma_bf16(a1, b, G.a[1][ni]);
    }
    __syncthreads();
    cur ^= 1;
  }
}

// Generic GEMM: C(MxN) = A(MxK fp32) * Bt(NxK bf16)^T + bias, ACT in {none,
// gelu}, optional residual.
template <int ACT, bool RES>
__global__ void gemm_kernel(const float* __restrict__ A,
                            const unsigned short* __restrict__ Bt,
                            const float* __restrict__ bias,
                            const float* __restrict__ res, float* __restrict__ C,
                            int M, int N, int K) {
  extern __shared__ char smem[];
  long bm = blockIdx.x, bn = blockIdx.y;
  int w = threadIdx.x >> 5, wm = w & 3, wn = w >> 2;
  int lane = threadIdx.x & 31, lr = lane & 15, hi = (lane >> 4) & 1;
  GemmAcc G;
  gemm_core(G, smem, A, Bt, K, bm, bn, wm, wn);
#pragma unroll
  for (int mi = 0; mi < 2; ++mi) {
#pragma unroll
    for (int r = 0; r < 8; ++r) {
      long m = bm * 128 + wm * 32 + mi * 16 + r + hi * 8;
      float* crow = C + m * N;
      const float* rrow = res + m * N;
#pragma unroll
      for (int ni = 0; ni < 4; ++ni) {
        long n = bn * 128 + wn * 64 + ni * 16 + lr;
        float v = G.a[mi][ni][r] + bias[n];
        if (ACT == 1) v = gelu_f(v);
        if (RES) v += rrow[n];
        crow[n] = v;
      }
    }
  }
}

// QKV GEMM: x(4096x512) * WqkvT(1536x512)^T + bqkv, scattered into bf16
// Q (b,h,t,dk) pre-scaled by 1/sqrt(dk), K (b,h,t,dk), V^T (b,h,dk,t).
__global__ void gemm_qkv_kernel(const float* __restrict__ A,
                                const unsigned short* __restrict__ Bt,
                                const float* __restrict__ bias,
                                unsigned short* __restrict__ q,
                                unsigned short* __restrict__ k,
                                unsigned short* __restrict__ vt) {
  extern __shared__ char smem[];
  long bm = blockIdx.x, bn = blockIdx.y;
  int w = threadIdx.x >> 5, wm = w & 3, wn = w >> 2;
  int lane = threadIdx.x & 31, lr = lane & 15, hi = (lane >> 4) & 1;
  GemmAcc G;
  gemm_core(G, smem, A, Bt, 512, bm, bn, wm, wn);
#pragma unroll
  for (int mi = 0; mi < 2; ++mi) {
#pragma unroll
    for (int r = 0; r < 8; ++r) {
      long m = bm * 128 + wm * 32 + mi * 16 + r + hi * 8;
      int b = (int)(m >> 9), t = (int)(m & 511);
#pragma unroll
      for (int ni = 0; ni < 4; ++ni) {
        long n = bn * 128 + wn * 64 + ni * 16 + lr;
        float v = G.a[mi][ni][r] + bias[n];
        int c = (int)(n >> 9);
        int rem = (int)(n & 511);
        int h = rem >> 6, dk = rem & 63;
        long bh = (long)b * 8 + h;
        if (c == 0)
          q[(bh * 512 + t) * 64 + dk] = f2bf(v * 0.125f);  // fold 1/sqrt(64)
        else if (c == 1)
          k[(bh * 512 + t) * 64 + dk] = f2bf(v);
        else
          vt[(bh * 64 + dk) * 512 + t] = f2bf(v);
      }
    }
  }
}

// ---------------------------------------------------------------------------
// Attention: one block per (b,h, 64-row t block).
// scores = Q*K^T + bias, mask, softmax, out = P*V. All matmuls via WMMA.
// Dynamic LDS layout (217088 B total, fits 320KB WGP):
//   sS: 64 x (512*4+16)  fp32 scores        132096 B
//   sP: 64 x 1040        bf16 probabilities  66560 B
//   sQ: 64 x 144         bf16 Q tile          9216 B
//   sT: 64 x 144         bf16 K/V^T staging   9216 B
// ---------------------------------------------------------------------------
__global__ void attn_kernel(const unsigned short* __restrict__ q,
                            const unsigned short* __restrict__ kk,
                            const unsigned short* __restrict__ vt,
                            const float* __restrict__ biasb,
                            const unsigned char* __restrict__ mask,
                            float* __restrict__ out) {
  extern __shared__ char smem[];
  char* sS = smem;
  char* sP = sS + 64 * 2064;
  char* sQ = sP + 64 * 1040;
  char* sT = sQ + 64 * 144;
  const float NEG_INF = -__builtin_inff();

  int bh = blockIdx.y;  // b*8 + h
  int b = bh >> 3, h = bh & 7;
  long tb = blockIdx.x;  // t block (0..7)
  int w = threadIdx.x >> 5, wm = w & 3, wn = w >> 2;
  int lane = threadIdx.x & 31, lr = lane & 15, hi = (lane >> 4) & 1;

  // Stage 64x64 Q tile (bf16) once.
  stage_bf16_64(sQ, 144, q, 64, (long)bh * 512 + tb * 64, 0);

  // ---- scores = Q * K^T (+bias, mask) ----
  for (int sc = 0; sc < 8; ++sc) {
    if (sc) __syncthreads();
    stage_bf16_64(sT, 144, kk, 64, (long)bh * 512 + sc * 64, 0);
    __syncthreads();
    v8f c0 = {};
    v8f c1 = {};
#pragma unroll
    for (int ks = 0; ks < 2; ++ks) {
      v16bf a = lds_frag(sQ, wm * 16, 144, ks * 64);
      v16bf b0 = lds_frag(sT, wn * 32, 144, ks * 64);
      v16bf b1 = lds_frag(sT, wn * 32 + 16, 144, ks * 64);
      c0 = wmma_bf16(a, b0, c0);
      c1 = wmma_bf16(a, b1, c1);
    }
#pragma unroll
    for (int r = 0; r < 8; ++r) {
      int tl = wm * 16 + r + hi * 8;
      long t = tb * 64 + tl;
      int s0 = sc * 64 + wn * 32 + lr;
      const float* brow = biasb + ((long)b * 512 + t) * 512;
      float v0 = c0[r] + brow[s0];
      float v1 = c1[r] + brow[s0 + 16];
      v0 = mask[b * 512 + s0] ? v0 : NEG_INF;
      v1 = mask[b * 512 + s0 + 16] ? v1 : NEG_INF;
      float* srow = (float*)(sS + tl * 2064);
      srow[s0] = v0;
      srow[s0 + 16] = v1;
    }
  }
  __syncthreads();

  // ---- softmax per row; write bf16 P ----
  for (int rr = 0; rr < 8; ++rr) {
    int row = w * 8 + rr;
    float* srow = (float*)(sS + row * 2064);
    float vals[16];
    float mx = NEG_INF;
#pragma unroll
    for (int j = 0; j < 16; ++j) {
      vals[j] = srow[lane + j * 32];
      mx = fmaxf(mx, vals[j]);
    }
#pragma unroll
    for (int off = 16; off >= 1; off >>= 1) mx = fmaxf(mx, __shfl_xor(mx, off, 32));
    float sum = 0.0f;
#pragma unroll
    for (int j = 0; j < 16; ++j) {
      float e = __expf(vals[j] - mx);
      vals[j] = e;
      sum += e;
    }
#pragma unroll
    for (int off = 16; off >= 1; off >>= 1) sum += __shfl_xor(sum, off, 32);
    float inv = 1.0f / sum;
    unsigned short* prow = (unsigned short*)(sP + row * 1040);
#pragma unroll
    for (int j = 0; j < 16; ++j) prow[lane + j * 32] = f2bf(vals[j] * inv);
  }
  __syncthreads();

  // ---- out = P * V  (B-operand = V^T rows are dk, contiguous over s) ----
  v8f c0 = {};
  v8f c1 = {};
  for (int ks2 = 0; ks2 < 8; ++ks2) {
    if (ks2) __syncthreads();
    stage_bf16_64(sT, 144, vt, 512, (long)bh * 64, ks2 * 64);
    __syncthreads();
#pragma unroll
    for (int ks = 0; ks < 2; ++ks) {
      v16bf a = lds_frag(sP, wm * 16, 1040, (ks2 * 2 + ks) * 64);
      v16bf b0 = lds_frag(sT, wn * 32, 144, ks * 64);
      v16bf b1 = lds_frag(sT, wn * 32 + 16, 144, ks * 64);
      c0 = wmma_bf16(a, b0, c0);
      c1 = wmma_bf16(a, b1, c1);
    }
  }
#pragma unroll
  for (int r = 0; r < 8; ++r) {
    int tl = wm * 16 + r + hi * 8;
    long t = tb * 64 + tl;
    int n0 = wn * 32 + lr;
    float* orow = out + ((long)b * 512 + t) * 512 + h * 64;
    orow[n0] = c0[r];
    orow[n0 + 16] = c1[r];
  }
}

// ---------------------------------------------------------------------------
// LayerNorm over D=512, one block per row, 256 threads (2 elems/thread).
// ---------------------------------------------------------------------------
__global__ void layernorm_kernel(const float* __restrict__ y,
                                 const float* __restrict__ g,
                                 const float* __restrict__ bb,
                                 float* __restrict__ out) {
  __shared__ float red[16];
  long row = blockIdx.x;
  const float* yr = y + row * 512;
  float a0 = yr[threadIdx.x], a1 = yr[threadIdx.x + 256];
  float s = a0 + a1, s2 = a0 * a0 + a1 * a1;
#pragma unroll
  for (int off = 16; off >= 1; off >>= 1) {
    s += __shfl_xor(s, off, 32);
    s2 += __shfl_xor(s2, off, 32);
  }
  int w = threadIdx.x >> 5, lane = threadIdx.x & 31;
  if (lane == 0) {
    red[w] = s;
    red[w + 8] = s2;
  }
  __syncthreads();
  float ts = 0.0f, ts2 = 0.0f;
#pragma unroll
  for (int i = 0; i < 8; ++i) {
    ts += red[i];
    ts2 += red[i + 8];
  }
  float mu = ts * (1.0f / 512.0f);
  float var = ts2 * (1.0f / 512.0f) - mu * mu;
  float inv = rsqrtf(var + 1e-5f);
  out[row * 512 + threadIdx.x] = (a0 - mu) * inv * g[threadIdx.x] + bb[threadIdx.x];
  out[row * 512 + threadIdx.x + 256] =
      (a1 - mu) * inv * g[threadIdx.x + 256] + bb[threadIdx.x + 256];
}

// ---------------------------------------------------------------------------
// Host orchestration.
// ---------------------------------------------------------------------------
extern "C" void kernel_launch(void* const* d_in, const int* in_sizes, int n_in,
                              void* d_out, int out_size, void* d_ws,
                              size_t ws_size, hipStream_t stream) {
  const float* x_in = (const float*)d_in[0];
  const unsigned char* mask = (const unsigned char*)d_in[1];
  const float* tmat = (const float*)d_in[2];
  const float* Wqkv = (const float*)d_in[3];
  const float* bqkv = (const float*)d_in[4];
  const float* Wproj = (const float*)d_in[5];
  const float* bproj = (const float*)d_in[6];
  const float* tw1 = (const float*)d_in[7];
  const float* tb1 = (const float*)d_in[8];
  const float* tw2 = (const float*)d_in[9];
  const float* tb2 = (const float*)d_in[10];
  const float* ln1g = (const float*)d_in[11];
  const float* ln1b = (const float*)d_in[12];
  const float* ln2g = (const float*)d_in[13];
  const float* ln2b = (const float*)d_in[14];
  const float* mw1 = (const float*)d_in[15];
  const float* mb1 = (const float*)d_in[16];
  const float* mw2 = (const float*)d_in[17];
  const float* mb2 = (const float*)d_in[18];

  char* ws = (char*)d_ws;
  // Workspace layout (bytes)
  const size_t WT_PER_LAYER = 6291456;               // bf16 weights per layer
  const size_t OFF_BIAS = 37748736;                  // (8,512,512) f32
  const size_t OFF_Q = OFF_BIAS + 8388608;           // (8,8,512,64) bf16
  const size_t OFF_K = OFF_Q + 4194304;
  const size_t OFF_VT = OFF_K + 4194304;             // (8,8,64,512) bf16
  const size_t OFF_ATTN = OFF_VT + 4194304;          // (4096,512) f32
  const size_t OFF_Y = OFF_ATTN + 8388608;           // (4096,512) f32
  const size_t OFF_X1 = OFF_Y + 8388608;             // (4096,512) f32
  const size_t OFF_H = OFF_X1 + 8388608;             // (4096,2048) f32
  const size_t OFF_XA = OFF_H + 33554432;
  const size_t OFF_XB = OFF_XA + 8388608;

  // --- one-time weight convert+transpose to bf16 (N x K layout) ---
  for (int l = 0; l < 6; ++l) {
    char* base = ws + (size_t)l * WT_PER_LAYER;
    for (int c = 0; c < 3; ++c)
      transpose_cvt_kernel<<<dim3(16, 16), 256, 0, stream>>>(
          Wqkv + ((size_t)l * 3 + c) * 262144,
          (unsigned short*)(base + (size_t)c * 524288), 512, 512);
    transpose_cvt_kernel<<<dim3(16, 16), 256, 0, stream>>>(
        Wproj + (size_t)l * 262144, (unsigned short*)(base + 1572864), 512, 512);
    transpose_cvt_kernel<<<dim3(16, 64), 256, 0, stream>>>(
        mw1 + (size_t)l * 1048576, (unsigned short*)(base + 2097152), 512, 2048);
    transpose_cvt_kernel<<<dim3(64, 16), 256, 0, stream>>>(
        mw2 + (size_t)l * 1048576, (unsigned short*)(base + 4194304), 2048, 512);
  }

  float* biasb = (float*)(ws + OFF_BIAS);
  unsigned short* q = (unsigned short*)(ws + OFF_Q);
  unsigned short* k = (unsigned short*)(ws + OFF_K);
  unsigned short* vt = (unsigned short*)(ws + OFF_VT);
  float* attn = (float*)(ws + OFF_ATTN);
  float* y = (float*)(ws + OFF_Y);
  float* x1 = (float*)(ws + OFF_X1);
  float* hb = (float*)(ws + OFF_H);
  float* xA = (float*)(ws + OFF_XA);
  float* xB = (float*)(ws + OFF_XB);

  const float* xcur = x_in;
  for (int l = 0; l < 6; ++l) {
    char* wbase = ws + (size_t)l * WT_PER_LAYER;
    const unsigned short* WqkvT = (const unsigned short*)wbase;
    const unsigned short* WprojT = (const unsigned short*)(wbase + 1572864);
    const unsigned short* mw1T = (const unsigned short*)(wbase + 2097152);
    const unsigned short* mw2T = (const unsigned short*)(wbase + 4194304);

    // temporal bias, shared across heads
    bias_mlp_kernel<<<8192, 256, 0, stream>>>(
        tmat, tw1 + (size_t)l * 64, tb1 + (size_t)l * 64, tw2 + (size_t)l * 64,
        tb2 + l, biasb, 2097152L);

    // QKV projection (scatter to Q/K/V^T bf16): M=4096, N=1536, K=512
    gemm_qkv_kernel<<<dim3(32, 12), 256, 40960, stream>>>(
        xcur, WqkvT, bqkv + (size_t)l * 1536, q, k, vt);

    // attention
    attn_kernel<<<dim3(8, 64), 256, 217088, stream>>>(q, k, vt, biasb, mask,
                                                      attn);

    // output projection + residual, then LN1: M=4096, N=512, K=512
    gemm_kernel<0, true><<<dim3(32, 4), 256, 40960, stream>>>(
        attn, WprojT, bproj + (size_t)l * 512, xcur, y, 4096, 512, 512);
    layernorm_kernel<<<4096, 256, 0, stream>>>(y, ln1g + (size_t)l * 512,
                                               ln1b + (size_t)l * 512, x1);

    // FFN: gelu(x1@mw1+mb1)@mw2+mb2 + x1, then LN2
    gemm_kernel<1, false><<<dim3(32, 16), 256, 40960, stream>>>(
        x1, mw1T, mb1 + (size_t)l * 2048, x1, hb, 4096, 2048, 512);
    gemm_kernel<0, true><<<dim3(32, 4), 256, 40960, stream>>>(
        hb, mw2T, mb2 + (size_t)l * 512, x1, y, 4096, 512, 2048);

    float* xnext = (l == 5) ? (float*)d_out : ((l & 1) ? xB : xA);
    layernorm_kernel<<<4096, 256, 0, stream>>>(y, ln2g + (size_t)l * 512,
                                               ln2b + (size_t)l * 512, xnext);
    xcur = xnext;
  }
  (void)in_sizes; (void)n_in; (void)out_size; (void)ws_size;
}
